// Decoder_28097676051300
// MI455X (gfx1250) — compile-verified, matrix operand order checked by reading, not run on hardware
//
#include <hip/hip_runtime.h>
#include <hip/hip_bf16.h>
#include <math.h>

// ---------------- problem constants ----------------
#define BSZ    64
#define TENC   480
#define EDIM   512
#define NMELS  80
#define TMELD  600
#define ARNNH  1024
#define PREH   256
#define ATTD   128
#define LOCF   32
#define LOCK   31
#define FRAME  160      // NMELS * 2
#define STEPS  300
#define PADW   15

typedef __attribute__((ext_vector_type(16))) _Float16 v16h;
typedef __attribute__((ext_vector_type(8)))  float    v8f;

// =====================================================================
// WMMA fragment helpers (CDNA5 16x16x32 f16, ISA 7.12.2 layouts)
// A (16x32 f16): lanes 0-15 -> M=lane, K in {0..7,16..23}; lanes 16-31 -> M=lane-16, K in {8..15,24..31}
// =====================================================================
__device__ __forceinline__ v16h load_a_frag(const _Float16* __restrict__ A, int lda,
                                            int mrow, int kt, int lane) {
  const _Float16* p = A + (size_t)mrow * lda + kt * 32 + ((lane >> 4) & 1) * 8;
  v16h a;
#pragma unroll
  for (int v = 0; v < 4; ++v) {
    a[2 * v]     = p[2 * v];
    a[2 * v + 1] = p[2 * v + 1];
    a[8 + 2 * v]     = p[16 + 2 * v];
    a[8 + 2 * v + 1] = p[16 + 2 * v + 1];
  }
  return a;
}

// Generic GEMM: C[M,N] = A[M,K](f16,row-major,lda) * B(pre-swizzled from W[N,K]) (+bias)
// one wave computes one 16x16 f32 tile.
__global__ void wmma_gemm(const _Float16* __restrict__ A, int lda,
                          const _Float16* __restrict__ Bsw,
                          float* __restrict__ C, int M, int N, int K,
                          const float* __restrict__ bias) {
  int wave = (blockIdx.x * blockDim.x + threadIdx.x) >> 5;
  int ntN = N >> 4;
  int total = (M >> 4) * ntN;
  if (wave >= total) return;
  int mt = wave / ntN, nt = wave % ntN;
  int lane = threadIdx.x & 31;
  int mrow = mt * 16 + (lane & 15);
  int nk = K >> 5;
  v8f acc = {};
  const _Float16* bp = Bsw + ((size_t)nt * nk) * 512 + lane * 16;
  for (int kt = 0; kt < nk; ++kt) {
    v16h a = load_a_frag(A, lda, mrow, kt, lane);
    v16h b = *(const v16h*)(bp + (size_t)kt * 512);
    acc = __builtin_amdgcn_wmma_f32_16x16x32_f16(false, a, false, b, (short)0, acc, false, false);
  }
  int ncol = nt * 16 + (lane & 15);
  int mbase = mt * 16 + ((lane >> 4) & 1) * 8;
  float bv = bias ? bias[ncol] : 0.0f;
#pragma unroll
  for (int v = 0; v < 8; ++v)
    C[(size_t)(mbase + v) * N + ncol] = acc[v] + bv;
}

// Fused dual GEMM for LSTM gates: C[64,N] = A1*B1 + A2*B2 (x*Wih^T + h*Whh^T)
__global__ void wmma_gemm_dual(const _Float16* __restrict__ A1, int lda1,
                               const _Float16* __restrict__ B1, int K1,
                               const _Float16* __restrict__ A2, int lda2,
                               const _Float16* __restrict__ B2, int K2,
                               float* __restrict__ C, int N) {
  int wave = (blockIdx.x * blockDim.x + threadIdx.x) >> 5;
  int ntN = N >> 4;
  int total = (BSZ >> 4) * ntN;
  if (wave >= total) return;
  int mt = wave / ntN, nt = wave % ntN;
  int lane = threadIdx.x & 31;
  int mrow = mt * 16 + (lane & 15);
  v8f acc = {};
  int nk1 = K1 >> 5;
  const _Float16* bp1 = B1 + ((size_t)nt * nk1) * 512 + lane * 16;
  for (int kt = 0; kt < nk1; ++kt) {
    v16h a = load_a_frag(A1, lda1, mrow, kt, lane);
    v16h b = *(const v16h*)(bp1 + (size_t)kt * 512);
    acc = __builtin_amdgcn_wmma_f32_16x16x32_f16(false, a, false, b, (short)0, acc, false, false);
  }
  int nk2 = K2 >> 5;
  const _Float16* bp2 = B2 + ((size_t)nt * nk2) * 512 + lane * 16;
  for (int kt = 0; kt < nk2; ++kt) {
    v16h a = load_a_frag(A2, lda2, mrow, kt, lane);
    v16h b = *(const v16h*)(bp2 + (size_t)kt * 512);
    acc = __builtin_amdgcn_wmma_f32_16x16x32_f16(false, a, false, b, (short)0, acc, false, false);
  }
  int ncol = nt * 16 + (lane & 15);
  int mbase = mt * 16 + ((lane >> 4) & 1) * 8;
#pragma unroll
  for (int v = 0; v < 8; ++v)
    C[(size_t)(mbase + v) * N + ncol] = acc[v];
}

// Swizzle W[N,K] f32 -> f16 B-fragments: frag (nt,kt) stored as 32 lanes x 16 halves.
__global__ void swizzle_w(const float* __restrict__ W, _Float16* __restrict__ Bsw,
                          int N, int K) {
  long i = (long)blockIdx.x * blockDim.x + threadIdx.x;
  long total = (long)N * K;
  if (i >= total) return;
  int fi = (int)(i >> 9);
  int r = (int)(i & 511);
  int lane = r >> 4, idx = r & 15;
  int nkt = K >> 5;
  int nt = fi / nkt, kt = fi % nkt;
  int nl = lane & 15;
  int v = idx >> 1, h = idx & 1;
  int kl = (v < 4 ? 2 * v : 16 + 2 * (v - 4)) + ((lane >> 4) & 1) * 8 + h;
  Bsw[i] = (_Float16)W[(size_t)(nt * 16 + nl) * K + kt * 32 + kl];
}

__global__ void f32_to_f16_kernel(const float* __restrict__ s, _Float16* __restrict__ d, long n) {
  long i = (long)blockIdx.x * blockDim.x + threadIdx.x;
  if (i < n) d[i] = (_Float16)s[i];
}

__global__ void zero_words(unsigned int* __restrict__ p, long n) {
  long i = (long)blockIdx.x * blockDim.x + threadIdx.x;
  if (i < n) p[i] = 0u;
}

// prenet: x_t -> relu(relu(x w1^T) w2^T), writes f16 into xa[:,0:256]
__global__ void prenet_kernel(const float* __restrict__ target,
                              const float* __restrict__ w1, const float* __restrict__ w2,
                              _Float16* __restrict__ xa, int step) {
  __shared__ float xs[FRAME];
  __shared__ float h1[PREH];
  int b = blockIdx.x, tid = threadIdx.x;            // 256 threads
  if (tid < FRAME) {
    float v = 0.0f;
    if (step > 0) {
      int m = tid % 80, u = 2 * (step - 1) + tid / 80;
      v = target[(size_t)b * NMELS * TMELD + (size_t)m * TMELD + u];
    }
    xs[tid] = v;
  }
  __syncthreads();
  float a = 0.0f;
  const float* w1r = w1 + (size_t)tid * FRAME;
  for (int k = 0; k < FRAME; ++k) a += xs[k] * w1r[k];
  h1[tid] = fmaxf(a, 0.0f);
  __syncthreads();
  float a2 = 0.0f;
  const float* w2r = w2 + (size_t)tid * PREH;
  for (int k = 0; k < PREH; ++k) a2 += h1[k] * w2r[k];
  xa[(size_t)b * (PREH + EDIM) + tid] = (_Float16)fmaxf(a2, 0.0f);
}

__global__ void lstm_pointwise(const float* __restrict__ gates,
                               const float* __restrict__ bih, const float* __restrict__ bhh,
                               float* __restrict__ c, _Float16* __restrict__ hdst, int ldh) {
  int i = blockIdx.x * blockDim.x + threadIdx.x;    // 64*1024
  if (i >= BSZ * ARNNH) return;
  int b = i >> 10, j = i & 1023;
  const float* g = gates + (size_t)b * 4096;
  float gi = g[j]          + bih[j]          + bhh[j];
  float gf = g[j + 1024]   + bih[j + 1024]   + bhh[j + 1024];
  float gg = g[j + 2048]   + bih[j + 2048]   + bhh[j + 2048];
  float go = g[j + 3072]   + bih[j + 3072]   + bhh[j + 3072];
  float si = 1.0f / (1.0f + __expf(-gi));
  float sf = 1.0f / (1.0f + __expf(-gf));
  float so = 1.0f / (1.0f + __expf(-go));
  float c2 = sf * c[i] + si * tanhf(gg);
  c[i] = c2;
  hdst[(size_t)b * ldh + j] = (_Float16)(so * tanhf(c2));
}

// location conv + tanh energy: one block (128 thr) per (b,t)
__global__ void energies_kernel(const float* __restrict__ aw, const float* __restrict__ awc,
                                const float* __restrict__ conv, const float* __restrict__ wloc,
                                const float* __restrict__ q, const float* __restrict__ pm,
                                const float* __restrict__ vvec, float* __restrict__ e) {
  __shared__ float loc[LOCF];
  __shared__ float red[128];
  int bt = blockIdx.x;
  int b = bt / TENC, t = bt % TENC;
  int tid = threadIdx.x;
  if (tid < LOCF) {
    float acc = 0.0f;
    const float* cw = conv + (size_t)tid * 2 * LOCK;
    for (int k = 0; k < LOCK; ++k) {
      int pos = t + k - PADW;
      if (pos >= 0 && pos < TENC) {
        acc += aw[b * TENC + pos] * cw[k] + awc[b * TENC + pos] * cw[LOCK + k];
      }
    }
    loc[tid] = acc;
  }
  __syncthreads();
  float s = q[b * ATTD + tid] + pm[(size_t)bt * ATTD + tid];
  const float* wl = wloc + (size_t)tid * LOCF;
#pragma unroll
  for (int f = 0; f < LOCF; ++f) s += wl[f] * loc[f];
  red[tid] = vvec[tid] * tanhf(s);
  __syncthreads();
  for (int st = 64; st > 0; st >>= 1) {
    if (tid < st) red[tid] += red[tid + st];
    __syncthreads();
  }
  if (tid == 0) e[bt] = red[0];
}

// softmax over T + context + state updates; one block (256 thr) per batch row
__global__ void softctx_kernel(const float* __restrict__ e, const float* __restrict__ memory,
                               float* __restrict__ aw, float* __restrict__ awc,
                               _Float16* __restrict__ xa, _Float16* __restrict__ xd,
                               _Float16* __restrict__ xp,
                               float* __restrict__ align_out, int step) {
  __shared__ float w[TENC];
  __shared__ float red[256];
  int b = blockIdx.x, tid = threadIdx.x;
  float m = -1e30f;
  for (int t = tid; t < TENC; t += 256) { float v = e[b * TENC + t]; w[t] = v; m = fmaxf(m, v); }
  red[tid] = m; __syncthreads();
  for (int st = 128; st > 0; st >>= 1) { if (tid < st) red[tid] = fmaxf(red[tid], red[tid + st]); __syncthreads(); }
  m = red[0]; __syncthreads();
  float s = 0.0f;
  for (int t = tid; t < TENC; t += 256) { float ex = __expf(w[t] - m); w[t] = ex; s += ex; }
  red[tid] = s; __syncthreads();
  for (int st = 128; st > 0; st >>= 1) { if (tid < st) red[tid] += red[tid + st]; __syncthreads(); }
  float inv = 1.0f / red[0]; __syncthreads();
  for (int t = tid; t < TENC; t += 256) {
    float ww = w[t] * inv; w[t] = ww;
    aw[b * TENC + t] = ww;
    awc[b * TENC + t] += ww;
    align_out[(size_t)b * STEPS * TENC + (size_t)step * TENC + t] = ww;
  }
  __syncthreads();
  for (int eo = tid; eo < EDIM; eo += 256) {
    float acc = 0.0f;
    const float* mp = memory + (size_t)b * TENC * EDIM + eo;
    for (int t = 0; t < TENC; ++t) acc += w[t] * mp[(size_t)t * EDIM];
    _Float16 h = (_Float16)acc;
    xa[(size_t)b * (PREH + EDIM) + PREH + eo] = h;
    xd[(size_t)b * (ARNNH + EDIM) + ARNNH + eo] = h;
    xp[(size_t)b * (ARNNH + EDIM) + ARNNH + eo] = h;
  }
}

__global__ void spec_scatter(const float* __restrict__ proj, float* __restrict__ spec, int step) {
  int i = blockIdx.x * blockDim.x + threadIdx.x;
  if (i >= BSZ * FRAME) return;
  int b = i / FRAME, f = i % FRAME;
  int m = f % 80, u = 2 * step + f / 80;
  spec[(size_t)b * NMELS * TMELD + (size_t)m * TMELD + u] = proj[i];
}

// =====================================================================
extern "C" void kernel_launch(void* const* d_in, const int* in_sizes, int n_in,
                              void* d_out, int out_size, void* d_ws, size_t ws_size,
                              hipStream_t stream) {
  (void)in_sizes; (void)n_in; (void)out_size; (void)ws_size;
  const float* memory    = (const float*)d_in[0];
  const float* target    = (const float*)d_in[1];
  const float* prenet_w1 = (const float*)d_in[2];
  const float* prenet_w2 = (const float*)d_in[3];
  const float* arnn_wih  = (const float*)d_in[4];
  const float* arnn_whh  = (const float*)d_in[5];
  const float* arnn_bih  = (const float*)d_in[6];
  const float* arnn_bhh  = (const float*)d_in[7];
  const float* att_wq    = (const float*)d_in[8];
  const float* att_wmem  = (const float*)d_in[9];
  const float* att_conv  = (const float*)d_in[10];
  const float* att_wloc  = (const float*)d_in[11];
  const float* att_v     = (const float*)d_in[12];
  const float* drnn_wih  = (const float*)d_in[13];
  const float* drnn_whh  = (const float*)d_in[14];
  const float* drnn_bih  = (const float*)d_in[15];
  const float* drnn_bhh  = (const float*)d_in[16];
  const float* proj_w    = (const float*)d_in[17];
  const float* proj_b    = (const float*)d_in[18];

  float* spec_out  = (float*)d_out;                       // [64,80,600]
  float* align_out = spec_out + (size_t)BSZ * NMELS * TMELD;

  // ---- carve workspace (256B aligned) ----
  char* ws = (char*)d_ws;
  size_t off = 0;
  auto carve = [&](size_t bytes) -> char* {
    off = (off + 255) & ~(size_t)255;
    char* p = ws + off;
    off += bytes;
    return p;
  };
  // zeroed state region
  size_t zstart_mark = (off + 255) & ~(size_t)255;
  float*    ac  = (float*)   carve((size_t)BSZ * ARNNH * 4);
  float*    dc  = (float*)   carve((size_t)BSZ * ARNNH * 4);
  float*    aw  = (float*)   carve((size_t)BSZ * TENC * 4);
  float*    awc = (float*)   carve((size_t)BSZ * TENC * 4);
  _Float16* xa  = (_Float16*)carve((size_t)BSZ * (PREH + EDIM) * 2);
  _Float16* xd  = (_Float16*)carve((size_t)BSZ * (ARNNH + EDIM) * 2);
  _Float16* xp  = (_Float16*)carve((size_t)BSZ * (ARNNH + EDIM) * 2);
  size_t zend_mark = off;
  // scratch
  float*    gates = (float*)   carve((size_t)BSZ * 4096 * 4);
  float*    qbuf  = (float*)   carve((size_t)BSZ * ATTD * 4);
  float*    ebuf  = (float*)   carve((size_t)BSZ * TENC * 4);
  float*    projo = (float*)   carve((size_t)BSZ * FRAME * 4);
  float*    pm    = (float*)   carve((size_t)BSZ * TENC * ATTD * 4);
  _Float16* mem16 = (_Float16*)carve((size_t)BSZ * TENC * EDIM * 2);
  _Float16* sw_awih = (_Float16*)carve((size_t)4096 * 768 * 2);
  _Float16* sw_awhh = (_Float16*)carve((size_t)4096 * 1024 * 2);
  _Float16* sw_dwih = (_Float16*)carve((size_t)4096 * 1536 * 2);
  _Float16* sw_dwhh = (_Float16*)carve((size_t)4096 * 1024 * 2);
  _Float16* sw_wq   = (_Float16*)carve((size_t)ATTD * 1024 * 2);
  _Float16* sw_wmem = (_Float16*)carve((size_t)ATTD * EDIM * 2);
  _Float16* sw_proj = (_Float16*)carve((size_t)FRAME * 1536 * 2);

  // ---- one-time (per launch) setup ----
  {
    long zwords = (long)((zend_mark - zstart_mark) / 4);
    zero_words<<<(unsigned)((zwords + 255) / 256), 256, 0, stream>>>(
        (unsigned int*)(ws + zstart_mark), zwords);
  }
  {
    long n = (long)BSZ * TENC * EDIM;
    f32_to_f16_kernel<<<(unsigned)((n + 255) / 256), 256, 0, stream>>>(memory, mem16, n);
  }
  auto swz = [&](const float* W, _Float16* dst, int N, int K) {
    long n = (long)N * K;
    swizzle_w<<<(unsigned)((n + 255) / 256), 256, 0, stream>>>(W, dst, N, K);
  };
  swz(arnn_wih, sw_awih, 4096, 768);
  swz(arnn_whh, sw_awhh, 4096, 1024);
  swz(drnn_wih, sw_dwih, 4096, 1536);
  swz(drnn_whh, sw_dwhh, 4096, 1024);
  swz(att_wq,   sw_wq,   ATTD, 1024);
  swz(att_wmem, sw_wmem, ATTD, EDIM);
  swz(proj_w,   sw_proj, FRAME, 1536);

  auto gemm_blocks = [](int M, int N) -> unsigned {
    int waves = (M / 16) * (N / 16);
    return (unsigned)((waves + 7) / 8);       // 8 waves (256 threads) per block
  };

  // processed_memory [B*T,128] = mem16 [B*T,512] * wmem^T  (WMMA)
  wmma_gemm<<<gemm_blocks(BSZ * TENC, ATTD), 256, 0, stream>>>(
      mem16, EDIM, sw_wmem, pm, BSZ * TENC, ATTD, EDIM, nullptr);

  // ---- 300 sequential decoder steps ----
  for (int s = 0; s < STEPS; ++s) {
    // 1. prenet -> xa[:,0:256]
    prenet_kernel<<<BSZ, 256, 0, stream>>>(target, prenet_w1, prenet_w2, xa, s);
    // 2. attention LSTM gates = xa*Wih^T + h(xd[:,0:1024])*Whh^T
    wmma_gemm_dual<<<gemm_blocks(BSZ, 4096), 256, 0, stream>>>(
        xa, PREH + EDIM, sw_awih, 768,
        xd, ARNNH + EDIM, sw_awhh, ARNNH, gates, 4096);
    // 3. pointwise -> ah (f16 into xd[:,0:1024]), ac
    lstm_pointwise<<<(BSZ * ARNNH) / 256, 256, 0, stream>>>(
        gates, arnn_bih, arnn_bhh, ac, xd, ARNNH + EDIM);
    // 4. q = ah * Wq^T
    wmma_gemm<<<gemm_blocks(BSZ, ATTD), 256, 0, stream>>>(
        xd, ARNNH + EDIM, sw_wq, qbuf, BSZ, ATTD, ARNNH, nullptr);
    // 5. energies
    energies_kernel<<<BSZ * TENC, 128, 0, stream>>>(
        aw, awc, att_conv, att_wloc, qbuf, pm, att_v, ebuf);
    // 6. softmax + context + alignment output
    softctx_kernel<<<BSZ, 256, 0, stream>>>(
        ebuf, memory, aw, awc, xa, xd, xp, align_out, s);
    // 7. decoder LSTM gates = [ah,ctx]*Wih^T + dh(xp[:,0:1024])*Whh^T
    wmma_gemm_dual<<<gemm_blocks(BSZ, 4096), 256, 0, stream>>>(
        xd, ARNNH + EDIM, sw_dwih, ARNNH + EDIM,
        xp, ARNNH + EDIM, sw_dwhh, ARNNH, gates, 4096);
    // 8. pointwise -> dh (f16 into xp[:,0:1024]), dc
    lstm_pointwise<<<(BSZ * ARNNH) / 256, 256, 0, stream>>>(
        gates, drnn_bih, drnn_bhh, dc, xp, ARNNH + EDIM);
    // 9. projection [dh,ctx] -> 160 (+bias)
    wmma_gemm<<<gemm_blocks(BSZ, FRAME), 256, 0, stream>>>(
        xp, ARNNH + EDIM, sw_proj, projo, BSZ, FRAME, ARNNH + EDIM, proj_b);
    // 10. scatter into [B,80,600] output layout
    spec_scatter<<<(BSZ * FRAME + 255) / 256, 256, 0, stream>>>(projo, spec_out, s);
  }
}
